// VectorQuantizer_19232863552080
// MI455X (gfx1250) — matrix operand order repaired and checked by reading, not compiled, not run on hardware
//
#include <hip/hip_runtime.h>
#include <hip/hip_bf16.h>

typedef _Float16 v16h __attribute__((ext_vector_type(16)));
typedef _Float16 v8h  __attribute__((ext_vector_type(8)));
typedef _Float16 v2h  __attribute__((ext_vector_type(2)));
typedef float    v8f  __attribute__((ext_vector_type(8)));

#define N_ROWS 131072
#define DIM    64
#define KCODES 1024
#define LDS_ROW_BYTES 144   // 64 halves (128B) + 16B pad -> 16 lanes hit 16 distinct banks

// ---------------------------------------------------------------------------
// Prep: f32 codebook -> f16 copy in ws, per-code squared norms, zero accums.
// One wave (32 lanes) per codebook row; lane handles 2 of the 64 dims.
// ---------------------------------------------------------------------------
__global__ void vq_prep(const float* __restrict__ emb, _Float16* __restrict__ embh,
                        float* __restrict__ enorm, unsigned* __restrict__ counts,
                        float* __restrict__ lossAcc) {
    const int gtid = blockIdx.x * blockDim.x + threadIdx.x;
    if (gtid < KCODES) counts[gtid] = 0u;
    if (gtid == 0)     *lossAcc = 0.0f;
    const int row  = gtid >> 5;
    const int lane = gtid & 31;
    if (row < KCODES) {
        const float2 v = *(const float2*)(emb + row * DIM + lane * 2);
        v2h h; h[0] = (_Float16)v.x; h[1] = (_Float16)v.y;
        *(v2h*)(embh + row * DIM + lane * 2) = h;
        float sq = v.x * v.x + v.y * v.y;
        #pragma unroll
        for (int m = 16; m >= 1; m >>= 1) sq += __shfl_xor(sq, m, 32);
        if (lane == 0) enorm[row] = sq;
    }
}

__device__ __forceinline__ void pack8(v16h& v, int base, float4 a, float4 b) {
    v[base + 0] = (_Float16)a.x; v[base + 1] = (_Float16)a.y;
    v[base + 2] = (_Float16)a.z; v[base + 3] = (_Float16)a.w;
    v[base + 4] = (_Float16)b.x; v[base + 5] = (_Float16)b.y;
    v[base + 6] = (_Float16)b.z; v[base + 7] = (_Float16)b.w;
}

// ---------------------------------------------------------------------------
// Main: 256 threads = 8 waves per WG; each wave owns 16 rows of z_e and scans
// all 1024 codes via v_wmma_f32_16x16x32_f16 (two per 16-code tile, K=64).
// Codebook (f16) + e-norms staged in LDS (148 KB of the 320 KB WGP pool).
// ---------------------------------------------------------------------------
__global__ void vq_main(const float* __restrict__ z_e, const float* __restrict__ emb,
                        const _Float16* __restrict__ embh, const float* __restrict__ enorm,
                        unsigned* __restrict__ counts, float* __restrict__ lossAcc,
                        float* __restrict__ out) {
    extern __shared__ char smem[];
    _Float16* ldsB = (_Float16*)smem;
    float*    ldsE = (float*)(smem + KCODES * LDS_ROW_BYTES);

    const int tid = threadIdx.x;

    // Stage f16 codebook into padded LDS rows (16B chunks, conflict-free pad).
    for (int r = tid; r < KCODES; r += 256) {
        const uint4* src = (const uint4*)(embh + r * DIM);
        #pragma unroll
        for (int j = 0; j < 8; ++j)
            *(uint4*)((char*)ldsB + r * LDS_ROW_BYTES + j * 16) = src[j];
    }
    *(float4*)(ldsE + tid * 4) = *(const float4*)(enorm + tid * 4);
    __syncthreads();

    const int wave = tid >> 5;
    const int lane = tid & 31;
    const int h    = lane >> 4;    // lane half (WMMA operand layout)
    const int ln   = lane & 15;
    const int rowBase = blockIdx.x * 128 + wave * 16;
    const int myRow   = rowBase + ln;

    // A operands: 16x32 f16, lane = row M; halves per ISA layout:
    // op1 covers K 0..31 (lane holds 8h..8h+7, 16+8h..), op2 covers K 32..63.
    v16h a1, a2;
    {
        const float* zr = z_e + (size_t)myRow * DIM;
        const int o = 8 * h;
        pack8(a1, 0, *(const float4*)(zr + o),      *(const float4*)(zr + o + 4));
        pack8(a1, 8, *(const float4*)(zr + 16 + o), *(const float4*)(zr + 16 + o + 4));
        pack8(a2, 0, *(const float4*)(zr + 32 + o), *(const float4*)(zr + 32 + o + 4));
        pack8(a2, 8, *(const float4*)(zr + 48 + o), *(const float4*)(zr + 48 + o + 4));
    }

    float best[8]; int bidx[8];
    #pragma unroll
    for (int r = 0; r < 8; ++r) { best[r] = 3.402823466e38f; bidx[r] = 0; }

    #pragma unroll 2
    for (int t = 0; t < 64; ++t) {
        const int code = t * 16 + ln;
        // B operand: lane = code column; dims 8h..8h+7, 16+8h..16+8h+7 (table order, matching A).
        const char* rowp = (const char*)ldsB + code * LDS_ROW_BYTES + h * 16;
        v8h b1lo = *(const v8h*)(rowp);
        v8h b1hi = *(const v8h*)(rowp + 32);
        v8h b2lo = *(const v8h*)(rowp + 64);
        v8h b2hi = *(const v8h*)(rowp + 96);
        v16h b1 = __builtin_shufflevector(b1lo, b1hi, 0,1,2,3,4,5,6,7,8,9,10,11,12,13,14,15);
        v16h b2 = __builtin_shufflevector(b2lo, b2hi, 0,1,2,3,4,5,6,7,8,9,10,11,12,13,14,15);

        v8f c = {};
        c = __builtin_amdgcn_wmma_f32_16x16x32_f16(false, a1, false, b1, (short)0, c, false, false);
        c = __builtin_amdgcn_wmma_f32_16x16x32_f16(false, a2, false, b2, (short)0, c, false, false);

        const float en = ldsE[code];
        // score = ||e||^2 - 2 z.e  (||z||^2 is constant per row -> irrelevant to argmin)
        #pragma unroll
        for (int r = 0; r < 8; ++r) {
            const float cand = fmaf(-2.0f, c[r], en);
            if (cand < best[r]) { best[r] = cand; bidx[r] = code; }
        }
    }

    // Butterfly min+argmin across the 16 lanes of each half (rows 0..7 / 8..15).
    #pragma unroll
    for (int r = 0; r < 8; ++r) {
        #pragma unroll
        for (int m = 1; m < 16; m <<= 1) {
            const float os = __shfl_xor(best[r], m, 32);
            const int   oi = __shfl_xor(bidx[r], m, 32);
            if (os < best[r] || (os == best[r] && oi < bidx[r])) { best[r] = os; bidx[r] = oi; }
        }
    }

    // Gather quantized rows, write straight-through output, loss + histogram.
    float lacc = 0.0f;
    #pragma unroll
    for (int m = 0; m < 16; ++m) {
        const int r   = m & 7;
        const int idx = __shfl(bidx[r], (m < 8) ? 0 : 16, 32);
        const int orow = rowBase + m;
        const float2 zv = *(const float2*)(z_e + (size_t)orow * DIM + lane * 2);
        const float2 qv = *(const float2*)(emb + (size_t)idx  * DIM + lane * 2);
        float* op = out + 1 + (size_t)orow * DIM + lane * 2;  // out[0] is loss -> 4B-aligned stores
        op[0] = qv.x; op[1] = qv.y;
        const float dx = qv.x - zv.x, dy = qv.y - zv.y;
        lacc = fmaf(dx, dx, lacc);
        lacc = fmaf(dy, dy, lacc);
        if (lane == 0) atomicAdd(&counts[idx], 1u);
    }
    #pragma unroll
    for (int m = 16; m >= 1; m >>= 1) lacc += __shfl_xor(lacc, m, 32);
    if (lane == 0) atomicAdd(lossAcc, lacc);
}

// ---------------------------------------------------------------------------
// Finalize: loss scalar + perplexity from the usage histogram. 1024 threads.
// ---------------------------------------------------------------------------
__global__ void vq_final(const unsigned* __restrict__ counts, const float* __restrict__ lossAcc,
                         float* __restrict__ out) {
    __shared__ float red[32];
    const int t = threadIdx.x;
    const float p = (float)counts[t] * (1.0f / (float)N_ROWS);
    float e = p * __logf(p + 1e-10f);
    #pragma unroll
    for (int m = 16; m >= 1; m >>= 1) e += __shfl_xor(e, m, 32);
    if ((t & 31) == 0) red[t >> 5] = e;
    __syncthreads();
    if (t < 32) {
        float v = red[t];
        #pragma unroll
        for (int m = 16; m >= 1; m >>= 1) v += __shfl_xor(v, m, 32);
        if (t == 0) {
            out[0] = 1.25f * lossAcc[0] * (1.0f / ((float)N_ROWS * (float)DIM));
            out[1 + (size_t)N_ROWS * DIM] = __expf(-v);
        }
    }
}

extern "C" void kernel_launch(void* const* d_in, const int* in_sizes, int n_in,
                              void* d_out, int out_size, void* d_ws, size_t ws_size,
                              hipStream_t stream) {
    const float* z_e = (const float*)d_in[0];
    const float* emb = (const float*)d_in[1];
    char* ws = (char*)d_ws;
    _Float16* embh   = (_Float16*)ws;                  // 131072 B  f16 codebook
    float*    enorm  = (float*)(ws + 131072);          //   4096 B  ||e_k||^2
    unsigned* counts = (unsigned*)(ws + 135168);       //   4096 B  histogram
    float*    lossAcc= (float*)(ws + 139264);          //      4 B  sq-err accumulator
    float* out = (float*)d_out;

    vq_prep<<<128, 256, 0, stream>>>(emb, embh, enorm, counts, lossAcc);

    const size_t lds_bytes = (size_t)KCODES * LDS_ROW_BYTES + KCODES * sizeof(float); // 151552
    vq_main<<<N_ROWS / 128, 256, lds_bytes, stream>>>(z_e, emb, embh, enorm, counts, lossAcc, out);

    vq_final<<<1, 1024, 0, stream>>>(counts, lossAcc, out);
}
